// PAM_Module_66408784331511
// MI455X (gfx1250) — compile-verified, matrix operand order checked by reading, not run on hardware
//
#include <hip/hip_runtime.h>
#include <hip/hip_bf16.h>
#include <math.h>

typedef _Float16 v8h  __attribute__((ext_vector_type(8)));
typedef _Float16 v16h __attribute__((ext_vector_type(16)));
typedef float    v8f  __attribute__((ext_vector_type(8)));

#define B_ 4
#define C_ 64
#define N_ 4096

// Build a CDNA5 16-bit WMMA fragment half (16 halves / lane) from two contiguous
// 16-byte chunks: elements [0..7] at p, [8..15] at p+16 halves. Caller applies the
// lane-dependent base offset so lanes 0-15 cover K {0..7,16..23} and lanes 16-31
// cover K {8..15,24..31} per the ISA 7.12.2 layout tables.
__device__ __forceinline__ v16h ld16h(const _Float16* p) {
  v8h lo = *(const v8h*)(p);
  v8h hi = *(const v8h*)(p + 16);
  return __builtin_shufflevector(lo, hi, 0,1,2,3,4,5,6,7,8,9,10,11,12,13,14,15);
}

__device__ __forceinline__ v8f wmma_f16(v16h a, v16h b, v8f c) {
  return __builtin_amdgcn_wmma_f32_16x16x32_f16(false, a, false, b, (short)0, c,
                                                false, false);
}

// DPP16 lane permute (VALU, no LDS round-trip). Controls used:
//   0xB1 = quad_perm(1,0,3,2)   : pair-swap within quads
//   0x4E = quad_perm(2,3,0,1)   : quad half-swap
//   0x141 = row_half_mirror     : mirror within 8 lanes (combines quads)
//   0x140 = row_mirror          : mirror within 16 lanes (combines octets)
// None cross the 16-lane boundary, so the two C/D half-wave row groups stay
// independent — exactly what the 16x16 WMMA D layout requires.
template <int CTRL>
__device__ __forceinline__ float dppf(float x) {
  return __builtin_bit_cast(
      float, __builtin_amdgcn_update_dpp(0, __builtin_bit_cast(int, x), CTRL,
                                         0xF, 0xF, true));
}
__device__ __forceinline__ float red16_max(float v) {
  v = fmaxf(v, dppf<0xB1>(v));
  v = fmaxf(v, dppf<0x4E>(v));
  v = fmaxf(v, dppf<0x141>(v));
  v = fmaxf(v, dppf<0x140>(v));
  return v;
}
__device__ __forceinline__ float red16_sum(float v) {
  v += dppf<0xB1>(v);
  v += dppf<0x4E>(v);
  v += dppf<0x141>(v);
  v += dppf<0x140>(v);
  return v;
}

// ---------------------------------------------------------------------------
// Kernel 1: channel-conv q/k projections (k=3, SAME over channel axis) and the
// 1x1 value conv, emitted as f16 for the WMMA pipeline.
//   qh/kh: [B][N][C] (c contiguous)   vh: [B][C][N] (n contiguous, bias folded)
// ---------------------------------------------------------------------------
__global__ __launch_bounds__(256) void pam_proj(
    const float* __restrict__ x, const float* __restrict__ qw,
    const float* __restrict__ kw, const float* __restrict__ vw,
    const float* __restrict__ vb,
    _Float16* __restrict__ qh, _Float16* __restrict__ kh,
    _Float16* __restrict__ vh) {
  __shared__ float xt[64][65];   // [c][n] padded -> conflict-free column reads
  __shared__ float vws[64][64];  // v_w [o][i]
  const int tid = threadIdx.x;
  const int b   = blockIdx.x >> 6;
  const int n0  = (blockIdx.x & 63) << 6;

  for (int i = 0; i < 16; ++i) {
    int idx = tid + 256 * i;                // 0..4095
    int c = idx >> 6, n = idx & 63;
    xt[c][n] = x[((size_t)(b * C_ + c)) * N_ + n0 + n];
  }
  for (int i = 0; i < 16; ++i) {
    int idx = tid + 256 * i;
    vws[idx >> 6][idx & 63] = vw[idx];
  }
  __syncthreads();

  const float q0 = qw[0], q1 = qw[1], q2 = qw[2];
  const float k0 = kw[0], k1 = kw[1], k2 = kw[2];

  // q/k: c fastest -> coalesced [n][c] f16 stores
  for (int i = 0; i < 16; ++i) {
    int idx = tid + 256 * i;
    int c = idx & 63, n = idx >> 6;
    float xm = (c > 0)  ? xt[c - 1][n] : 0.f;
    float xc = xt[c][n];
    float xp = (c < 63) ? xt[c + 1][n] : 0.f;
    size_t o = ((size_t)(b * N_ + n0 + n)) * C_ + c;
    qh[o] = (_Float16)(q0 * xm + q1 * xc + q2 * xp);
    kh[o] = (_Float16)(k0 * xm + k1 * xc + k2 * xp);
  }
  // v: n fastest -> coalesced [c][n] f16 stores; vws row broadcast across lanes
  for (int i = 0; i < 16; ++i) {
    int idx = tid + 256 * i;
    int n = idx & 63, co = idx >> 6;
    float acc = vb[co];
#pragma unroll
    for (int ci = 0; ci < 64; ++ci) acc += vws[co][ci] * xt[ci][n];
    vh[((size_t)(b * C_ + co)) * N_ + n0 + n] = (_Float16)acc;
  }
}

// ---------------------------------------------------------------------------
// Kernel 2: fused flash attention. 4 waves/block, each wave owns 16 rows n;
// the block streams 64-wide m tiles of k and v^T through LDS. Per m-tile,
// each wave does 8 WMMAs for S = q k^T (16x64, K=64), online softmax with
// DPP16 butterfly reductions, then 8 WMMAs for O += P V^T (16x64 f32 accum).
// Writes y = gamma * softmax(qk^T) v^T   (pre-BN) as f32 [B][C][N].
// ---------------------------------------------------------------------------
__global__ __launch_bounds__(128) void pam_attn(
    const _Float16* __restrict__ qh, const _Float16* __restrict__ kh,
    const _Float16* __restrict__ vh, const float* __restrict__ gamma,
    float* __restrict__ y) {
  __shared__ __align__(16) _Float16 kt[64 * 64];      // [m_local][c]   8 KB
  __shared__ __align__(16) _Float16 vt[64 * 64];      // [c][m_local]   8 KB
  __shared__ __align__(16) _Float16 pt[4][16 * 64];   // per-wave P     8 KB

  const int tid   = threadIdx.x;
  const int lane  = tid & 31;
  const int wave  = tid >> 5;
  const int b     = blockIdx.x >> 6;
  const int n0    = (blockIdx.x & 63) << 6;
  const int nbase = n0 + wave * 16;
  const int lrow  = lane & 15;
  const int off0  = (lane & 16) ? 8 : 0;    // ISA half-wave K offset
  const int rbase = (lane & 16) ? 8 : 0;    // C/D rows for this half-wave

  // A fragments of q (rows nbase..nbase+15, K chunks 0 and 32), kept in VGPRs.
  const _Float16* qrow = qh + ((size_t)(b * N_ + nbase + lrow)) * C_;
  const v16h qa0 = ld16h(qrow + 0  + off0);
  const v16h qa1 = ld16h(qrow + 32 + off0);

  v8f O0 = {}, O1 = {}, O2 = {}, O3 = {};
  float mrun[8], lrun[8];
#pragma unroll
  for (int r = 0; r < 8; ++r) { mrun[r] = -INFINITY; lrun[r] = 0.f; }

  for (int mi = 0; mi < N_ / 64; ++mi) {
    const int m0 = mi * 64;
    __syncthreads();  // previous iteration's LDS consumers done
    {   // k tile: 64 rows x 128 B, fully coalesced b128 (512 chunks)
      const uint4* src = (const uint4*)(kh + ((size_t)(b * N_ + m0)) * C_);
      uint4* dst = (uint4*)kt;
#pragma unroll
      for (int j = 0; j < 4; ++j) dst[tid + 128 * j] = src[tid + 128 * j];
    }
    {   // v^T tile: 64 rows (c) x 64 halves (m), rows contiguous in global
#pragma unroll
      for (int j = 0; j < 4; ++j) {
        int q = tid + 128 * j;
        int cc = q >> 3, pp = q & 7;
        *(uint4*)(vt + cc * 64 + pp * 8) =
            *(const uint4*)(vh + ((size_t)(b * C_ + cc)) * N_ + m0 + pp * 8);
      }
    }
    __syncthreads();

    // S[16x64] = q[16x64] * k^T : four 16x16 D tiles, K=64 -> 2 WMMAs each
    v8f S0 = {}, S1 = {}, S2 = {}, S3 = {};
    S0 = wmma_f16(qa0, ld16h(kt + (0  + lrow) * C_ + 0  + off0), S0);
    S0 = wmma_f16(qa1, ld16h(kt + (0  + lrow) * C_ + 32 + off0), S0);
    S1 = wmma_f16(qa0, ld16h(kt + (16 + lrow) * C_ + 0  + off0), S1);
    S1 = wmma_f16(qa1, ld16h(kt + (16 + lrow) * C_ + 32 + off0), S1);
    S2 = wmma_f16(qa0, ld16h(kt + (32 + lrow) * C_ + 0  + off0), S2);
    S2 = wmma_f16(qa1, ld16h(kt + (32 + lrow) * C_ + 32 + off0), S2);
    S3 = wmma_f16(qa0, ld16h(kt + (48 + lrow) * C_ + 0  + off0), S3);
    S3 = wmma_f16(qa1, ld16h(kt + (48 + lrow) * C_ + 32 + off0), S3);

    _Float16* prow = &pt[wave][0];
    float sc[8];
#pragma unroll
    for (int r = 0; r < 8; ++r) {
      // row max over 64 columns: local max of 4 tiles, then DPP16 butterfly
      float v = fmaxf(fmaxf(S0[r], S1[r]), fmaxf(S2[r], S3[r]));
      v = red16_max(v);
      float mnew = fmaxf(mrun[r], v);
      float p0 = __expf(S0[r] - mnew);
      float p1 = __expf(S1[r] - mnew);
      float p2 = __expf(S2[r] - mnew);
      float p3 = __expf(S3[r] - mnew);
      sc[r]   = __expf(mrun[r] - mnew);
      mrun[r] = mnew;
      int row = r + rbase;                        // D-layout row for this lane
      prow[row * 64 + lrow]      = (_Float16)p0;  // stage P for A-layout reload
      prow[row * 64 + 16 + lrow] = (_Float16)p1;
      prow[row * 64 + 32 + lrow] = (_Float16)p2;
      prow[row * 64 + 48 + lrow] = (_Float16)p3;
      float s = red16_sum((p0 + p1) + (p2 + p3));
      lrun[r] = lrun[r] * sc[r] + s;
    }
#pragma unroll
    for (int r = 0; r < 8; ++r) {
      O0[r] *= sc[r]; O1[r] *= sc[r]; O2[r] *= sc[r]; O3[r] *= sc[r];
    }

    // O[16x64] += P[16x64] * V^T[64x64] : 8 WMMAs (per-wave LDS staging; DS
    // ops from the same wave are processed in order)
    const v16h pa0 = ld16h(prow + lrow * 64 + 0  + off0);
    const v16h pa1 = ld16h(prow + lrow * 64 + 32 + off0);
    O0 = wmma_f16(pa0, ld16h(vt + (0  + lrow) * 64 + 0  + off0), O0);
    O0 = wmma_f16(pa1, ld16h(vt + (0  + lrow) * 64 + 32 + off0), O0);
    O1 = wmma_f16(pa0, ld16h(vt + (16 + lrow) * 64 + 0  + off0), O1);
    O1 = wmma_f16(pa1, ld16h(vt + (16 + lrow) * 64 + 32 + off0), O1);
    O2 = wmma_f16(pa0, ld16h(vt + (32 + lrow) * 64 + 0  + off0), O2);
    O2 = wmma_f16(pa1, ld16h(vt + (32 + lrow) * 64 + 32 + off0), O2);
    O3 = wmma_f16(pa0, ld16h(vt + (48 + lrow) * 64 + 0  + off0), O3);
    O3 = wmma_f16(pa1, ld16h(vt + (48 + lrow) * 64 + 32 + off0), O3);
  }

  const float g = gamma[0];
#pragma unroll
  for (int r = 0; r < 8; ++r) {
    float inv = g / lrun[r];
    int n = nbase + r + rbase;
    y[((size_t)(b * C_ + ( 0 + lrow))) * N_ + n] = O0[r] * inv;
    y[((size_t)(b * C_ + (16 + lrow))) * N_ + n] = O1[r] * inv;
    y[((size_t)(b * C_ + (32 + lrow))) * N_ + n] = O2[r] * inv;
    y[((size_t)(b * C_ + (48 + lrow))) * N_ + n] = O3[r] * inv;
  }
}

// ---------------------------------------------------------------------------
// Kernel 3: per-channel batch stats over (B,H,W); fold BN into scale/shift.
// ---------------------------------------------------------------------------
__global__ __launch_bounds__(256) void pam_bnstats(
    const float* __restrict__ y, const float* __restrict__ bnw,
    const float* __restrict__ bnb, float* __restrict__ stats) {
  __shared__ float rs[256], rs2[256];
  const int c = blockIdx.x;
  const int tid = threadIdx.x;
  float s = 0.f, s2 = 0.f;
  for (int b = 0; b < B_; ++b) {
    const float* p = y + ((size_t)(b * C_ + c)) * N_;
    for (int n = tid; n < N_; n += 256) { float v = p[n]; s += v; s2 += v * v; }
  }
  rs[tid] = s; rs2[tid] = s2;
  __syncthreads();
  for (int off = 128; off > 0; off >>= 1) {
    if (tid < off) { rs[tid] += rs[tid + off]; rs2[tid] += rs2[tid + off]; }
    __syncthreads();
  }
  if (tid == 0) {
    const float inv = 1.f / (float)(B_ * N_);
    float mean = rs[0] * inv;
    float var  = rs2[0] * inv - mean * mean;
    float a = bnw[c] * rsqrtf(var + 1e-5f);
    stats[c]      = a;
    stats[C_ + c] = bnb[c] - mean * a;
  }
}

// ---------------------------------------------------------------------------
// Kernel 4: out = y*scale[c] + shift[c] + x   (BN affine + residual)
// ---------------------------------------------------------------------------
__global__ __launch_bounds__(256) void pam_final(
    const float* __restrict__ y, const float* __restrict__ x,
    const float* __restrict__ stats, float* __restrict__ out) {
  size_t i = (size_t)blockIdx.x * 256 + threadIdx.x;
  int c = (int)((i >> 12) & 63);   // layout [B][C][N], N = 4096
  out[i] = y[i] * stats[c] + stats[C_ + c] + x[i];
}

extern "C" void kernel_launch(void* const* d_in, const int* in_sizes, int n_in,
                              void* d_out, int out_size, void* d_ws, size_t ws_size,
                              hipStream_t stream) {
  const float* x     = (const float*)d_in[0];
  const float* qw    = (const float*)d_in[1];
  const float* kw    = (const float*)d_in[2];
  const float* vw    = (const float*)d_in[3];
  const float* vb    = (const float*)d_in[4];
  const float* gamma = (const float*)d_in[5];
  const float* bnw   = (const float*)d_in[6];
  const float* bnb   = (const float*)d_in[7];
  float* out = (float*)d_out;

  char* ws = (char*)d_ws;
  _Float16* qh    = (_Float16*)(ws);                       // 2 MB
  _Float16* kh    = (_Float16*)(ws + (2ull << 20));        // 2 MB
  _Float16* vh    = (_Float16*)(ws + (4ull << 20));        // 2 MB
  float*    ybuf  = (float*)   (ws + (6ull << 20));        // 4 MB
  float*    stats = (float*)   (ws + (10ull << 20));       // 512 B

  pam_proj   <<<B_ * (N_ / 64), 256, 0, stream>>>(x, qw, kw, vw, vb, qh, kh, vh);
  pam_attn   <<<B_ * (N_ / 64), 128, 0, stream>>>(qh, kh, vh, gamma, ybuf);
  pam_bnstats<<<C_,             256, 0, stream>>>(ybuf, bnw, bnb, stats);
  pam_final  <<<(B_ * C_ * N_) / 256, 256, 0, stream>>>(ybuf, x, stats, out);
}
